// Attention_v4_21603685499615
// MI455X (gfx1250) — matrix operand order repaired and checked by reading, not compiled, not run on hardware
//
#include <hip/hip_runtime.h>
#include <hip/hip_bf16.h>

// Problem constants (fixed by the reference)
#define BB   2
#define NN   512
#define AAT  14
#define DIMM 1024
#define HH   16
#define DD   64
#define MTOT (BB * NN * AAT)   // 14336
#define NQKV (3 * DIMM)        // 3072
#define NKSTEPS (DIMM / 32)    // 32 K-steps per GEMM

typedef __attribute__((ext_vector_type(16))) __bf16 v16bf;
typedef __attribute__((ext_vector_type(8)))  __bf16 v8bf;
typedef __attribute__((ext_vector_type(8)))  float  v8f;
typedef __attribute__((ext_vector_type(4)))  int    v4i;

union V16 { v16bf v; v8bf h[2]; };

// ---- gfx1250 async-copy feature detection (compile-safe fallback) ----------
#if defined(__has_builtin)
#  if __has_builtin(__builtin_amdgcn_global_load_async_to_lds_b128)
#    define HAVE_ASYNC_COPY 1
#  endif
#  if __has_builtin(__builtin_amdgcn_s_wait_asynccnt)
#    define HAVE_WAIT_ASYNC_BUILTIN 1
#  endif
#endif

#if defined(HAVE_ASYNC_COPY)
// Builtin signature (from compiler diagnostic): arg0 is int4 addrspace(1)* (non-const)
typedef __attribute__((address_space(1))) v4i* g4_p;
typedef __attribute__((address_space(3))) v4i* l4_p;
#  if defined(HAVE_WAIT_ASYNC_BUILTIN)
#    define WAIT_ASYNC_1() __builtin_amdgcn_s_wait_asynccnt(1)
#    define WAIT_ASYNC_0() __builtin_amdgcn_s_wait_asynccnt(0)
#  else
#    define WAIT_ASYNC_1() asm volatile("s_wait_asynccnt 0x1" ::: "memory")
#    define WAIT_ASYNC_0() asm volatile("s_wait_asynccnt 0x0" ::: "memory")
#  endif
#endif

// Stage one 16-byte chunk global -> LDS (async engine if available)
static __device__ __forceinline__ void stage16(const __bf16* g, __bf16* l) {
#if defined(HAVE_ASYNC_COPY)
    // generic LDS pointers carry the LDS byte offset in addr[31:0] (aperture rule);
    // global generic pointers are numerically identical to AS1.
    __builtin_amdgcn_global_load_async_to_lds_b128(
        (g4_p)(unsigned long long)(const void*)g,
        (l4_p)(unsigned int)(unsigned long long)(void*)l,
        0, 0);
#else
    *(v8bf*)l = *(const v8bf*)g;
#endif
}

static __device__ __forceinline__ unsigned short f2bf_bits(float f) {
    union { float f; unsigned u; } in; in.f = f;
    unsigned r = in.u + 0x7FFFu + ((in.u >> 16) & 1u);  // round-to-nearest-even
    return (unsigned short)(r >> 16);
}
static __device__ __forceinline__ __bf16 f2bf(float f) {
    union { unsigned short s; __bf16 b; } o; o.s = f2bf_bits(f);
    return o.b;
}

// ---------------------------------------------------------------------------
// Conversion kernels
// ---------------------------------------------------------------------------
__global__ __launch_bounds__(256) void k_cvt_x(const float4* __restrict__ x,
                                               unsigned long long* __restrict__ xb) {
    long i = (long)blockIdx.x * blockDim.x + threadIdx.x;
    float4 v = x[i];
    unsigned long long p = (unsigned long long)f2bf_bits(v.x)
                         | ((unsigned long long)f2bf_bits(v.y) << 16)
                         | ((unsigned long long)f2bf_bits(v.z) << 32)
                         | ((unsigned long long)f2bf_bits(v.w) << 48);
    xb[i] = p;
}

__global__ __launch_bounds__(256) void k_transpose_w(const float* __restrict__ w,
                                                     __bf16* __restrict__ wt,
                                                     int K, int Nc) {
    long i = (long)blockIdx.x * blockDim.x + threadIdx.x;
    if (i >= (long)K * Nc) return;
    int k = (int)(i / Nc), n = (int)(i % Nc);
    wt[(long)n * K + k] = f2bf(w[i]);   // wt[n][k] = w[k][n]
}

// ---------------------------------------------------------------------------
// Pipelined bf16 GEMM core (block: 256 thr = 8 waves; tile 128M x 64N, K-step 32)
//  - B tile staged in LDS (double buffered, shared by all waves, async copy)
//  - A operand register double-buffered
// LDS B tile row stride = 40 elements (80B): the 16-lane B-operand ds_loads hit
// 16 distinct 4-bank groups -> conflict-free.
// ---------------------------------------------------------------------------
#define BROW 40

#if defined(HAVE_ASYNC_COPY)
#  define WAIT_CUR(kt) do { if ((kt) + 1 < NKSTEPS) { WAIT_ASYNC_1(); } else { WAIT_ASYNC_0(); } } while (0)
#else
#  define WAIT_CUR(kt) do { } while (0)   /* ds_store visibility handled by __syncthreads */
#endif

#define GEMM_CORE(X_, Wt_, ACC_)                                                   \
    __shared__ __bf16 Bs[2][64][BROW];                                             \
    const int wave = threadIdx.x >> 5;                                             \
    const int lane = threadIdx.x & 31;                                             \
    const int hi = lane >> 4, ln = lane & 15;                                      \
    const int m0 = blockIdx.x * 128 + wave * 16;                                   \
    const int n0 = blockIdx.y * 64;                                                \
    const int srow = threadIdx.x >> 2;      /* 0..63  : staged B row (n)   */      \
    const int schk = threadIdx.x & 3;       /* 0..3   : 16B chunk within k */      \
    v8f ACC_[4] = {};                                                              \
    const __bf16* xrow = (X_) + (long)(m0 + ln) * DIMM;                            \
    const __bf16* wrow = (Wt_) + (long)(n0 + srow) * DIMM + schk * 8;              \
    /* prologue: stage tile 0, load A regs for k0 = 0 */                           \
    stage16(wrow, &Bs[0][srow][schk * 8]);                                         \
    V16 a_cur, a_nxt;                                                              \
    a_cur.h[0] = *(const v8bf*)(xrow + hi * 16);                                   \
    a_cur.h[1] = *(const v8bf*)(xrow + hi * 16 + 8);                               \
    for (int kt = 0; kt < NKSTEPS; ++kt) {                                         \
        const int k0 = kt * 32;                                                    \
        const int cur = kt & 1;                                                    \
        if (kt + 1 < NKSTEPS) {                                                    \
            stage16(wrow + k0 + 32, &Bs[cur ^ 1][srow][schk * 8]);                 \
            a_nxt.h[0] = *(const v8bf*)(xrow + k0 + 32 + hi * 16);                 \
            a_nxt.h[1] = *(const v8bf*)(xrow + k0 + 32 + hi * 16 + 8);             \
            __builtin_prefetch(xrow + k0 + 256, 0, 1);                             \
        }                                                                          \
        /* wait for current tile (keep next tile in flight), then cross-wave */    \
        WAIT_CUR(kt);                                                              \
        __syncthreads();                                                           \
        V16 b0, b1, b2, b3;                                                        \
        b0.h[0] = *(const v8bf*)&Bs[cur][0  + ln][hi * 16];                        \
        b0.h[1] = *(const v8bf*)&Bs[cur][0  + ln][hi * 16 + 8];                    \
        b1.h[0] = *(const v8bf*)&Bs[cur][16 + ln][hi * 16];                        \
        b1.h[1] = *(const v8bf*)&Bs[cur][16 + ln][hi * 16 + 8];                    \
        b2.h[0] = *(const v8bf*)&Bs[cur][32 + ln][hi * 16];                        \
        b2.h[1] = *(const v8bf*)&Bs[cur][32 + ln][hi * 16 + 8];                    \
        b3.h[0] = *(const v8bf*)&Bs[cur][48 + ln][hi * 16];                        \
        b3.h[1] = *(const v8bf*)&Bs[cur][48 + ln][hi * 16 + 8];                    \
        ACC_[0] = __builtin_amdgcn_wmma_f32_16x16x32_bf16(false, a_cur.v, false, b0.v, (short)0, ACC_[0], false, false); \
        ACC_[1] = __builtin_amdgcn_wmma_f32_16x16x32_bf16(false, a_cur.v, false, b1.v, (short)0, ACC_[1], false, false); \
        ACC_[2] = __builtin_amdgcn_wmma_f32_16x16x32_bf16(false, a_cur.v, false, b2.v, (short)0, ACC_[2], false, false); \
        ACC_[3] = __builtin_amdgcn_wmma_f32_16x16x32_bf16(false, a_cur.v, false, b3.v, (short)0, ACC_[3], false, false); \
        __syncthreads();  /* reads done before buffer is overwritten next iter */  \
        a_cur = a_nxt;                                                             \
    }

// ---------------------------------------------------------------------------
// GEMM 1: qkv = Xbf @ WqkvT^T + b_qkv, scatter epilogue into Q/K (scaled) and V^T
// ---------------------------------------------------------------------------
__global__ __launch_bounds__(256) void k_gemm_qkv(const __bf16* __restrict__ X,
                                                  const __bf16* __restrict__ Wt,
                                                  const float*  __restrict__ bias,
                                                  __bf16* __restrict__ Qb,
                                                  __bf16* __restrict__ Kb,
                                                  __bf16* __restrict__ Vt) {
    GEMM_CORE(X, Wt, acc)
    const float scale = 0.5946035575013605f;  // 64^-0.125 (2^-0.75), applied to both q and k
#pragma unroll
    for (int c = 0; c < 4; ++c) {
#pragma unroll
        for (int v = 0; v < 8; ++v) {
            int m = m0 + v + 8 * hi;          // C layout: row = vgpr + 8*(lane>=16)
            int n = n0 + c * 16 + ln;         //           col = lane & 15
            float val = acc[c][v] + bias[n];
            int b  = m / (NN * AAT);
            int rm = m % (NN * AAT);
            int i  = rm / AAT;
            int r  = rm % AAT;
            int h  = n / (3 * DD);
            int t  = (n / DD) % 3;            // 0=q 1=k 2=v
            int d  = n % DD;
            long inst = (long)(b * AAT + r) * HH + h;
            if (t == 0)      Qb[(inst * NN + i) * DD + d] = f2bf(val * scale);
            else if (t == 1) Kb[(inst * NN + i) * DD + d] = f2bf(val * scale);
            else             Vt[(inst * DD + d) * NN + i] = f2bf(val);  // V transposed
        }
    }
}

// ---------------------------------------------------------------------------
// GEMM 2: out = Attn @ WprojT^T + b_proj (fp32 out)
// ---------------------------------------------------------------------------
__global__ __launch_bounds__(256) void k_gemm_proj(const __bf16* __restrict__ X,
                                                   const __bf16* __restrict__ Wt,
                                                   const float*  __restrict__ bias,
                                                   float* __restrict__ Out) {
    GEMM_CORE(X, Wt, acc)
#pragma unroll
    for (int c = 0; c < 4; ++c) {
#pragma unroll
        for (int v = 0; v < 8; ++v) {
            int m = m0 + v + 8 * hi;
            int n = n0 + c * 16 + ln;
            Out[(long)m * DIMM + n] = acc[c][v] + bias[n];
        }
    }
}

// ---------------------------------------------------------------------------
// Flash attention: one wave handles 16 query rows of one (b,r,h) instance.
// V loads issued early; next K tile prefetched before the softmax VALU block.
// ---------------------------------------------------------------------------
__global__ __launch_bounds__(32) void k_attn(const __bf16* __restrict__ Qb,
                                             const __bf16* __restrict__ Kb,
                                             const __bf16* __restrict__ Vt,
                                             __bf16* __restrict__ O) {
    __shared__ unsigned short Pld[16][40];  // 16 x 32 bf16 P tile (+pad)

    const int inst = blockIdx.x;            // (b*A + r)*H + h
    const int it   = blockIdx.y;            // i-tile
    const int lane = threadIdx.x & 31;
    const int hi = lane >> 4, ln = lane & 15;

    const __bf16* Qi    = Qb + ((long)inst * NN + it * 16) * DD;
    const __bf16* Kbase = Kb + (long)inst * NN * DD;
    const __bf16* Vbase = Vt + (long)inst * DD * NN;

    // Q tile as two A-operands (d = 0..31 and 32..63), resident all loop
    V16 qa0, qa1;
    {
        const __bf16* qrow = Qi + ln * DD;
        qa0.h[0] = *(const v8bf*)(qrow + hi * 16);
        qa0.h[1] = *(const v8bf*)(qrow + hi * 16 + 8);
        qa1.h[0] = *(const v8bf*)(qrow + 32 + hi * 16);
        qa1.h[1] = *(const v8bf*)(qrow + 32 + hi * 16 + 8);
    }

    v8f acc[4] = {};
    float mrow[8], lrow[8];
#pragma unroll
    for (int v = 0; v < 8; ++v) { mrow[v] = -1e30f; lrow[v] = 0.0f; }

    // prefetch K operands for jt = 0
    v16bf kc0, kc1, kc2, kc3, kn0, kn1, kn2, kn3;
    {
        const __bf16* kr0 = Kbase + (long)(0 + ln) * DD + hi * 16;
        const __bf16* kr1 = Kbase + (long)(16 + ln) * DD + hi * 16;
        kc0 = *(const v16bf*)(kr0);
        kc1 = *(const v16bf*)(kr0 + 32);
        kc2 = *(const v16bf*)(kr1);
        kc3 = *(const v16bf*)(kr1 + 32);
    }

    for (int jt = 0; jt < NN; jt += 32) {
        // ---- V operands issued early (consumed after softmax) ----
        v16bf vb0 = *(const v16bf*)(Vbase + (long)(0  + ln) * NN + jt + hi * 16);
        v16bf vb1 = *(const v16bf*)(Vbase + (long)(16 + ln) * NN + jt + hi * 16);
        v16bf vb2 = *(const v16bf*)(Vbase + (long)(32 + ln) * NN + jt + hi * 16);
        v16bf vb3 = *(const v16bf*)(Vbase + (long)(48 + ln) * NN + jt + hi * 16);

        // ---- scores: two 16x16 tiles over d = 64 ----
        v8f s0 = {}, s1 = {};
        s0 = __builtin_amdgcn_wmma_f32_16x16x32_bf16(false, qa0.v, false, kc0, (short)0, s0, false, false);
        s0 = __builtin_amdgcn_wmma_f32_16x16x32_bf16(false, qa1.v, false, kc1, (short)0, s0, false, false);
        s1 = __builtin_amdgcn_wmma_f32_16x16x32_bf16(false, qa0.v, false, kc2, (short)0, s1, false, false);
        s1 = __builtin_amdgcn_wmma_f32_16x16x32_bf16(false, qa1.v, false, kc3, (short)0, s1, false, false);

        // ---- prefetch next K tile during the softmax VALU block ----
        if (jt + 32 < NN) {
            const __bf16* kr0 = Kbase + (long)(jt + 32 + ln) * DD + hi * 16;
            const __bf16* kr1 = Kbase + (long)(jt + 48 + ln) * DD + hi * 16;
            kn0 = *(const v16bf*)(kr0);
            kn1 = *(const v16bf*)(kr0 + 32);
            kn2 = *(const v16bf*)(kr1);
            kn3 = *(const v16bf*)(kr1 + 32);
        }

        // ---- online softmax (rows live inside 16-lane halves) ----
#pragma unroll
        for (int v = 0; v < 8; ++v) {
            float x = fmaxf(s0[v], s1[v]);
            x = fmaxf(x, __shfl_xor(x, 1, 32));
            x = fmaxf(x, __shfl_xor(x, 2, 32));
            x = fmaxf(x, __shfl_xor(x, 4, 32));
            x = fmaxf(x, __shfl_xor(x, 8, 32));
            float nm = fmaxf(mrow[v], x);
            float corr = __expf(mrow[v] - nm);
            mrow[v] = nm;
            float p0 = __expf(s0[v] - nm);
            float p1 = __expf(s1[v] - nm);
            float rs = p0 + p1;
            rs += __shfl_xor(rs, 1, 32);
            rs += __shfl_xor(rs, 2, 32);
            rs += __shfl_xor(rs, 4, 32);
            rs += __shfl_xor(rs, 8, 32);
            lrow[v] = lrow[v] * corr + rs;
#pragma unroll
            for (int c = 0; c < 4; ++c) acc[c][v] *= corr;
            int row = v + 8 * hi;
            Pld[row][ln]      = f2bf_bits(p0);
            Pld[row][16 + ln] = f2bf_bits(p1);
        }
        __syncthreads();

        // ---- P (16x32) as A-operand from LDS; V^T rows as B-operands ----
        V16 pa;
        pa.h[0] = *(const v8bf*)&Pld[ln][hi * 16];
        pa.h[1] = *(const v8bf*)&Pld[ln][hi * 16 + 8];
        acc[0] = __builtin_amdgcn_wmma_f32_16x16x32_bf16(false, pa.v, false, vb0, (short)0, acc[0], false, false);
        acc[1] = __builtin_amdgcn_wmma_f32_16x16x32_bf16(false, pa.v, false, vb1, (short)0, acc[1], false, false);
        acc[2] = __builtin_amdgcn_wmma_f32_16x16x32_bf16(false, pa.v, false, vb2, (short)0, acc[2], false, false);
        acc[3] = __builtin_amdgcn_wmma_f32_16x16x32_bf16(false, pa.v, false, vb3, (short)0, acc[3], false, false);
        __syncthreads();

        kc0 = kn0; kc1 = kn1; kc2 = kn2; kc3 = kn3;
    }

    // ---- epilogue: out[b, i, r, h*64+d] = acc / l ----
    const int h  = inst % HH;
    const int br = inst / HH;
    const int r  = br % AAT;
    const int b  = br / AAT;
#pragma unroll
    for (int c = 0; c < 4; ++c) {
#pragma unroll
        for (int v = 0; v < 8; ++v) {
            int i = it * 16 + v + 8 * hi;
            float o = acc[c][v] / lrow[v];
            long dst = (((long)(b * NN + i) * AAT + r) * DIMM) + h * DD + c * 16 + ln;
            O[dst] = f2bf(o);
        }
    }
}

// ---------------------------------------------------------------------------
extern "C" void kernel_launch(void* const* d_in, const int* in_sizes, int n_in,
                              void* d_out, int out_size, void* d_ws, size_t ws_size,
                              hipStream_t stream) {
    (void)in_sizes; (void)n_in; (void)out_size; (void)ws_size;
    const float* x      = (const float*)d_in[0];
    const float* w_qkv  = (const float*)d_in[1];
    const float* b_qkv  = (const float*)d_in[2];
    const float* w_proj = (const float*)d_in[3];
    const float* b_proj = (const float*)d_in[4];

    char* ws = (char*)d_ws;
    size_t off = 0;
    __bf16* Xb     = (__bf16*)(ws + off); off += (size_t)MTOT * DIMM * 2;          // 29.4 MB
    __bf16* WqkvT  = (__bf16*)(ws + off); off += (size_t)DIMM * NQKV * 2;          //  6.3 MB
    __bf16* WprojT = (__bf16*)(ws + off); off += (size_t)DIMM * DIMM * 2;          //  2.1 MB
    __bf16* Qb     = (__bf16*)(ws + off); off += (size_t)BB * AAT * HH * NN * DD * 2;
    __bf16* Kb     = (__bf16*)(ws + off); off += (size_t)BB * AAT * HH * NN * DD * 2;
    __bf16* Vt     = (__bf16*)(ws + off); off += (size_t)BB * AAT * HH * NN * DD * 2;
    __bf16* Attn   = (__bf16*)(ws + off); off += (size_t)MTOT * DIMM * 2;

    k_cvt_x<<<(MTOT * DIMM / 4) / 256, 256, 0, stream>>>((const float4*)x,
                                                         (unsigned long long*)Xb);
    k_transpose_w<<<(DIMM * NQKV + 255) / 256, 256, 0, stream>>>(w_qkv, WqkvT, DIMM, NQKV);
    k_transpose_w<<<(DIMM * DIMM + 255) / 256, 256, 0, stream>>>(w_proj, WprojT, DIMM, DIMM);

    k_gemm_qkv<<<dim3(MTOT / 128, NQKV / 64), 256, 0, stream>>>(Xb, WqkvT, b_qkv, Qb, Kb, Vt);

    k_attn<<<dim3(BB * AAT * HH, NN / 16), 32, 0, stream>>>(Qb, Kb, Vt, Attn);

    k_gemm_proj<<<dim3(MTOT / 128, DIMM / 64), 256, 0, stream>>>(Attn, WprojT, b_proj, (float*)d_out);
}
